// TopkGate_3951369912666
// MI455X (gfx1250) — compile-verified, matrix operand order checked by reading, not run on hardware
//
#include <hip/hip_runtime.h>

// Problem constants (from reference setup_inputs):
#define T_TOKENS 4096
#define PROTO    2
#define NE       32                  // experts per prototype
#define NEXP     (PROTO * NE)        // 64 total experts
#define KDIM     2048
#define CAP      160                 // ceil(1.25 * 4096 / 32)
#define TPAIRS   (T_TOKENS * PROTO)  // 8192 (token, prototype) pairs
#define SLICE    (NE * CAP)          // 5120 elements per (t,p) combine slice
#define TOTALCOMB ((size_t)TPAIRS * SLICE)  // 41,943,040 elems per output tensor

typedef __attribute__((ext_vector_type(16))) _Float16 v16h;
typedef __attribute__((ext_vector_type(8)))  float    v8f;
typedef __attribute__((ext_vector_type(4)))  float    v4f;

// ---------------------------------------------------------------------------
// Kernel 1: logits = X (4096x2048) @ W^T (2048x64), f16 WMMA w/ f32 accum.
// One 16x16 C tile per wave; 8 waves per 256-thread block; 1024 tiles total.
// A (16x32 f16) layout per ISA: lane<16 -> M=lane, halves {K0..7, K16..23};
// lane>=16 -> M=lane-16, halves {K8..15, K24..31}. B mirrors A since W is
// stored N-major (B^T row-major).
// ---------------------------------------------------------------------------
__global__ __launch_bounds__(256) void gemm_logits_wmma(
    const float* __restrict__ X, const float* __restrict__ Wm,
    float* __restrict__ logits) {
  const int lane   = threadIdx.x & 31;
  const int wid    = blockIdx.x * 8 + (threadIdx.x >> 5);  // 0..1023
  const int mt     = wid >> 2;     // 0..255 token tile
  const int nt     = wid & 3;      // 0..3 expert tile
  const int half16 = lane >> 4;    // which half-wave
  const int lane16 = lane & 15;

  const float* xrow = X  + (size_t)(mt * 16 + lane16) * KDIM;
  const float* wrow = Wm + (size_t)(nt * 16 + lane16) * KDIM;

  v8f c = {};
  for (int k0 = 0; k0 < KDIM; k0 += 32) {
    const int kA = k0 + half16 * 8;
    v16h a, b;
#pragma unroll
    for (int j = 0; j < 8; ++j) {
      a[j]     = (_Float16)xrow[kA + j];
      a[j + 8] = (_Float16)xrow[kA + 16 + j];
      b[j]     = (_Float16)wrow[kA + j];
      b[j + 8] = (_Float16)wrow[kA + 16 + j];
    }
    c = __builtin_amdgcn_wmma_f32_16x16x32_f16(
        /*neg_a=*/false, a, /*neg_b=*/false, b,
        /*c_mod=*/(short)0, c, /*reuse_a=*/false, /*reuse_b=*/false);
  }
  // C/D layout: VGPR v -> (M = v + 8*half16, N = lane16)
#pragma unroll
  for (int v = 0; v < 8; ++v) {
    const int mo = mt * 16 + v + half16 * 8;
    const int no = nt * 16 + lane16;
    logits[(size_t)mo * NEXP + no] = c[v];
  }
}

// ---------------------------------------------------------------------------
// Kernel 2: per-(token,prototype) softmax over 32 experts, top-1 and the
// best distinct runner-up (argmax tie-break = first index, like jnp.argmax).
// ---------------------------------------------------------------------------
__global__ __launch_bounds__(256) void softmax_top2(
    const float* __restrict__ logits, float* __restrict__ gates,
    int* __restrict__ idx1, int* __restrict__ idx2,
    float* __restrict__ g1, float* __restrict__ g2) {
  const int tp = blockIdx.x * blockDim.x + threadIdx.x;  // (t*P + p)
  if (tp >= TPAIRS) return;
  const float* lg = logits + (size_t)tp * NE;
  float v[NE];
  float mx = -INFINITY;
#pragma unroll
  for (int e = 0; e < NE; ++e) { v[e] = lg[e]; mx = fmaxf(mx, v[e]); }
  float s = 0.0f;
#pragma unroll
  for (int e = 0; e < NE; ++e) { v[e] = __expf(v[e] - mx); s += v[e]; }
  const float inv = 1.0f / s;
  int i1 = 0; float b1 = -1.0f;
#pragma unroll
  for (int e = 0; e < NE; ++e) {
    v[e] *= inv;
    if (v[e] > b1) { b1 = v[e]; i1 = e; }
  }
  int i2 = 0; float b2 = -1.0f;
#pragma unroll
  for (int e = 0; e < NE; ++e) {
    if (e != i1 && v[e] > b2) { b2 = v[e]; i2 = e; }
  }
  float* go = gates + (size_t)tp * NE;
#pragma unroll
  for (int e = 0; e < NE; ++e) go[e] = v[e];
  idx1[tp] = i1; idx2[tp] = i2; g1[tp] = b1; g2[tp] = b2;
}

// ---------------------------------------------------------------------------
// Kernel 3: round-1 expert-queue positions. One wave32 per (p,e); ballot +
// popcount prefix preserves token order. Writes loc1 (-1 if over capacity)
// and ce = uncapped top-1 count (used for l_aux and round-2 offset).
// ---------------------------------------------------------------------------
__global__ __launch_bounds__(32) void scan_round1(
    const int* __restrict__ idx1, int* __restrict__ loc1,
    int* __restrict__ ce) {
  const int pe = blockIdx.x;          // 0..63
  const int p = pe >> 5, e = pe & 31;
  const int lane = threadIdx.x;
  const unsigned lmask = (1u << lane) - 1u;
  int cnt = 0;
  for (int t0 = 0; t0 < T_TOKENS; t0 += 32) {
    const int tp = (t0 + lane) * PROTO + p;
    const bool m = (idx1[tp] == e);
    const unsigned bal = (unsigned)__ballot(m);
    if (m) {
      const int pos = cnt + __popc(bal & lmask);
      loc1[tp] = (pos < CAP) ? pos : -1;
    }
    cnt += __popc(bal);
  }
  if (lane == 0) ce[pe] = cnt;
}

// ---------------------------------------------------------------------------
// Kernel 4: round-2 positions. Reference detail: mask1 is capacity-capped
// BEFORE masking logits, so a token whose top-1 was dropped re-selects the
// SAME expert in round 2 (and can never fit, but still consumes cumsum
// order). Round-2 positions start at min(ce, CAP).
// ---------------------------------------------------------------------------
__global__ __launch_bounds__(32) void scan_round2(
    const int* __restrict__ idx1, const int* __restrict__ idx2,
    const int* __restrict__ loc1, const int* __restrict__ ce,
    int* __restrict__ loc2) {
  const int pe = blockIdx.x;
  const int p = pe >> 5, e = pe & 31;
  const int lane = threadIdx.x;
  const unsigned lmask = (1u << lane) - 1u;
  int off = ce[pe]; if (off > CAP) off = CAP;
  int cnt = 0;
  for (int t0 = 0; t0 < T_TOKENS; t0 += 32) {
    const int tp = (t0 + lane) * PROTO + p;
    const int sel = (loc1[tp] >= 0) ? idx2[tp] : idx1[tp];
    const bool m = (sel == e);
    const unsigned bal = (unsigned)__ballot(m);
    if (m) {
      const int pos = off + cnt + __popc(bal & lmask);
      loc2[tp] = (pos < CAP) ? pos : -1;
    }
    cnt += __popc(bal);
  }
}

// ---------------------------------------------------------------------------
// Kernel 5: l_aux = mean_{p,e}( mean_t(gates) * mean_t(mask1_uncapped) ).
// ---------------------------------------------------------------------------
__global__ __launch_bounds__(64) void laux_kernel(
    const float* __restrict__ gates, const int* __restrict__ ce,
    float* __restrict__ out) {
  __shared__ float red[NEXP];
  const int j = threadIdx.x;  // (p*32 + e)
  float s = 0.0f;
  for (int t = 0; t < T_TOKENS; ++t) s += gates[(size_t)t * NEXP + j];
  const float me  = s / (float)T_TOKENS;
  const float cef = (float)ce[j] / (float)T_TOKENS;
  red[j] = me * cef;
  __syncthreads();
  for (int w = NEXP / 2; w > 0; w >>= 1) {
    if (j < w) red[j] += red[j + w];
    __syncthreads();
  }
  if (j == 0) out[0] = red[0] / (float)NEXP;
}

// ---------------------------------------------------------------------------
// Kernel 6: materialize combine (f32) and dispatch_mask (0/1 f32). This is
// the HBM-store-bound part (~336 MB total > 192 MB L2), so:
//  - stores are non-temporal (don't thrash L2),
//  - bulk is b128: output starts at d_out+1 float, so each 5120-float slice
//    is 3 floats short of 16B alignment -> peel 3-float head + 1-float tail,
//    vectorize the 1279 aligned 16B chunks in the middle.
// ---------------------------------------------------------------------------
__global__ __launch_bounds__(256) void fill_outputs(
    const int* __restrict__ idx1, const int* __restrict__ idx2,
    const float* __restrict__ g1, const float* __restrict__ g2,
    const int* __restrict__ loc1, const int* __restrict__ loc2,
    float* __restrict__ out) {
  const int tp = blockIdx.x;  // 0..8191
  const int i1 = idx1[tp], i2 = idx2[tp];
  const int l1 = loc1[tp], l2 = loc2[tp];
  const float s1 = (l1 >= 0) ? g1[tp] : 0.0f;
  const float s2 = (l2 >= 0) ? g2[tp] : 0.0f;
  const float denom = fmaxf(s1 + s2, 1.1920929e-07f);  // finfo(f32).eps
  const float w1 = s1 / denom;
  const float w2 = s2 / denom;
  const int tgt1 = (l1 >= 0) ? i1 * CAP + l1 : -1;
  const int tgt2 = (l2 >= 0) ? i2 * CAP + l2 : -1;
  float* comb = out + 1 + (size_t)tp * SLICE;
  float* disp = out + 1 + TOTALCOMB + (size_t)tp * SLICE;

  // Middle: q in [3, 5119) as 1279 16B-aligned v4f chunks, NT stores.
  for (int i = threadIdx.x; i < (SLICE - 4) / 4; i += 256) {  // 1279 chunks
    const int q0 = 3 + 4 * i;
    v4f cv = {0.0f, 0.0f, 0.0f, 0.0f};
    v4f dv = {0.0f, 0.0f, 0.0f, 0.0f};
    if (tgt1 >= q0 && tgt1 < q0 + 4) { cv[tgt1 - q0] = w1;  dv[tgt1 - q0] = 1.0f; }
    if (tgt2 >= q0 && tgt2 < q0 + 4) { cv[tgt2 - q0] += w2; dv[tgt2 - q0] = 1.0f; }
    __builtin_nontemporal_store(cv, (v4f*)(comb + q0));
    __builtin_nontemporal_store(dv, (v4f*)(disp + q0));
  }

  // Head q=0..2 and tail q=5119 (scalar).
  if (threadIdx.x < 4) {
    const int q = (threadIdx.x < 3) ? (int)threadIdx.x : (SLICE - 1);
    float cvs = 0.0f, dvs = 0.0f;
    if (q == tgt1) { cvs = w1;  dvs = 1.0f; }
    if (q == tgt2) { cvs += w2; dvs = 1.0f; }
    __builtin_nontemporal_store(cvs, comb + q);
    __builtin_nontemporal_store(dvs, disp + q);
  }
}

extern "C" void kernel_launch(void* const* d_in, const int* in_sizes, int n_in,
                              void* d_out, int out_size, void* d_ws, size_t ws_size,
                              hipStream_t stream) {
  (void)in_sizes; (void)n_in; (void)out_size; (void)ws_size;
  const float* X  = (const float*)d_in[0];   // (4096, 2048) f32
  const float* Wm = (const float*)d_in[1];   // (64, 2048) f32
  float* out = (float*)d_out;                // [l_aux | combine | dispatch]

  // Workspace carve-up (~2.3 MB)
  float* logits = (float*)d_ws;                        // 4096*64
  float* gates  = logits + (size_t)T_TOKENS * NEXP;    // 4096*64
  int*   idx1   = (int*)(gates + (size_t)T_TOKENS * NEXP);
  int*   idx2   = idx1 + TPAIRS;
  float* g1     = (float*)(idx2 + TPAIRS);
  float* g2     = g1 + TPAIRS;
  int*   loc1   = (int*)(g2 + TPAIRS);
  int*   loc2   = loc1 + TPAIRS;
  int*   ce     = loc2 + TPAIRS;                       // 64 ints

  gemm_logits_wmma<<<128, 256, 0, stream>>>(X, Wm, logits);
  softmax_top2<<<TPAIRS / 256, 256, 0, stream>>>(logits, gates, idx1, idx2, g1, g2);
  scan_round1<<<NEXP, 32, 0, stream>>>(idx1, loc1, ce);
  scan_round2<<<NEXP, 32, 0, stream>>>(idx1, idx2, loc1, ce, loc2);
  laux_kernel<<<1, 64, 0, stream>>>(gates, ce, out);
  fill_outputs<<<TPAIRS, 256, 0, stream>>>(idx1, idx2, g1, g2, loc1, loc2, out);
}
